// PixelShuffleUpsampleV2_18064632447171
// MI455X (gfx1250) — matrix operand order repaired and checked by reading, not compiled, not run on hardware
//
#include <hip/hip_runtime.h>
#include <hip/hip_bf16.h>

typedef _Float16 v16h __attribute__((ext_vector_type(16)));
typedef _Float16 v8h  __attribute__((ext_vector_type(8)));
typedef float    v8f  __attribute__((ext_vector_type(8)));

// Problem constants
#define BB   2
#define CIN  64
#define HH   256
#define WW   256
#define HP   258        // padded
#define WP   258
#define COUT 256
#define NPIX (BB*HH*WW)             // 131072
#define XPAD_HALVES ((size_t)BB*HP*WP*CIN)   // 8,520,192
#define T_HALVES    ((size_t)BB*HH*WW*CIN)   // 8,388,608 per branch

// Workspace byte offsets (all 256B aligned)
#define XPAD_OFF   ((size_t)0)
#define T_OFF      ((size_t)17040384)
#define WB3_OFF    ((size_t)50594816)   // 2 branches * 4 nt * 18 chunks * 32 lanes * 16 halves
#define WB1_OFF    ((size_t)50742272)   // 2 branches * 16 nt * 2 chunks * 32 lanes * 16 halves
#define OFFB_OFF   ((size_t)50807808)   // fp32 NHWC (B,H,W,256)
#define SCOPE_OFF  ((size_t)185025536)  // f16  NHWC (B,H,W,256)

union AFrag { v16h v; v8h h[2]; };

// ---------------- Stage 0: zero padded input buffer ----------------
__global__ void k_zero_xpad(uint4* __restrict__ p, int n16) {
    int i = blockIdx.x * blockDim.x + threadIdx.x;
    if (i < n16) p[i] = uint4{0u, 0u, 0u, 0u};
}

// ---------------- Stage 1: x fp32 NCHW -> f16 NHWC with 1px halo ----------------
__global__ void k_pack_x(const float* __restrict__ x, _Float16* __restrict__ xpad) {
    int idx = blockIdx.x * blockDim.x + threadIdx.x;   // over B*CIN*H*W
    if (idx >= BB*CIN*HH*WW) return;
    int w = idx & 255, h = (idx >> 8) & 255, c = (idx >> 16) & 63, b = idx >> 22;
    xpad[(((size_t)(b*HP + h + 1))*WP + (w + 1))*CIN + c] = (_Float16)x[idx];
}

// ---------------- Stage 2: pack weights into per-lane WMMA B-fragment layout --------
// B-frag layout (16-bit B, 32x16): col N = lane&15, element j <-> K = 8*(lane>>4) + (j&7) + 16*(j>>3).
// K ordering for 3x3: K = tap*64 + cin  (tap = ky*3+kx), chunk = tap*2 + (cin>=32).
__global__ void k_pack_w(const float* __restrict__ ow1, const float* __restrict__ sw1,
                         const float* __restrict__ ow2, const float* __restrict__ sw2,
                         _Float16* __restrict__ wb3, _Float16* __restrict__ wb1) {
    int idx = blockIdx.x * blockDim.x + threadIdx.x;
    if (idx < 73728) {  // wb3: (((br*4+nt)*18+chunk)*32+lane)*16+j
        int branch = idx / 36864; int r = idx % 36864;
        int nt = r / 9216; r %= 9216;
        int chunk = r / 512; r %= 512;
        int lane = r / 16, j = r % 16;
        int K = chunk*32 + 8*(lane >> 4) + (j & 7) + 16*(j >> 3);
        int tap = K >> 6, cin = K & 63;
        int cout = nt*16 + (lane & 15);
        const float* wsrc = branch ? sw1 : ow1;   // (64,64,3,3)
        wb3[idx] = (_Float16)wsrc[(cout*64 + cin)*9 + tap];
    } else if (idx < 73728 + 32768) {  // wb1
        int id2 = idx - 73728;
        int branch = id2 / 16384; int r = id2 % 16384;
        int nt = r / 1024; r %= 1024;
        int chunk = r / 512; r %= 512;
        int lane = r / 16, j = r % 16;
        int cin = chunk*32 + 8*(lane >> 4) + (j & 7) + 16*(j >> 3);
        int cout = nt*16 + (lane & 15);
        const float* wsrc = branch ? sw2 : ow2;   // (256,64,1,1)
        wb1[id2] = (_Float16)wsrc[cout*64 + cin];
    }
}

// ---------------- Stage 3: 3x3 conv 64->64 + bias + ReLU (both branches), WMMA ------
// Wave: (2 M-tiles = 32 pixels along W) x (branch). 4 N-tiles, K = 18 chunks of 32.
// Each B fragment feeds 2 WMMAs (one per M-tile) to halve B traffic and raise the
// compute/load ratio per wait.
__global__ __launch_bounds__(256) void k_conv3(const _Float16* __restrict__ xpad,
                                               const _Float16* __restrict__ wb3,
                                               const float* __restrict__ ob1,
                                               const float* __restrict__ sb1,
                                               _Float16* __restrict__ t) {
    int lane = threadIdx.x & 31, wave = threadIdx.x >> 5;
    int pairIdx = blockIdx.x * 4 + (wave >> 1);   // 4 tile-pairs per block
    int branch  = wave & 1;
    int pix0 = pairIdx << 5;                      // 32 consecutive pixels (same row)
    int b = pix0 >> 16, h = (pix0 >> 8) & 255, w0 = pix0 & 255;
    int row = lane & 15, kh = lane >> 4;

    v8f acc[2][4] = {};
    const _Float16* wbase = wb3 + (size_t)branch * 36864;

    #pragma unroll
    for (int tap = 0; tap < 9; ++tap) {
        const _Float16* src0 = xpad +
            ((((size_t)b*HP + (h + tap/3))*WP) + (w0 + row + tap%3))*CIN;
        const _Float16* src1 = src0 + 16*CIN;
        #pragma unroll
        for (int cc = 0; cc < 2; ++cc) {
            AFrag a0, a1;
            a0.h[0] = *(const v8h*)(src0 + cc*32 +      kh*8);
            a0.h[1] = *(const v8h*)(src0 + cc*32 + 16 + kh*8);
            a1.h[0] = *(const v8h*)(src1 + cc*32 +      kh*8);
            a1.h[1] = *(const v8h*)(src1 + cc*32 + 16 + kh*8);
            int chunk = tap*2 + cc;
            const _Float16* wc = wbase + ((size_t)chunk*32 + lane)*16;
            #pragma unroll
            for (int nt = 0; nt < 4; ++nt) {
                v16h bf = *(const v16h*)(wc + (size_t)nt*9216);
                acc[0][nt] = __builtin_amdgcn_wmma_f32_16x16x32_f16(
                    false, a0.v, false, bf, (short)0, acc[0][nt], false, false);
                acc[1][nt] = __builtin_amdgcn_wmma_f32_16x16x32_f16(
                    false, a1.v, false, bf, (short)0, acc[1][nt], false, false);
            }
        }
    }

    const float* bias = branch ? sb1 : ob1;
    _Float16* tb = t + (size_t)branch * T_HALVES;
    #pragma unroll
    for (int m = 0; m < 2; ++m) {
        #pragma unroll
        for (int nt = 0; nt < 4; ++nt) {
            int ch = nt*16 + (lane & 15);
            float bv = bias[ch];
            #pragma unroll
            for (int r = 0; r < 8; ++r) {
                int p = pix0 + m*16 + r + 8*kh;
                float v = acc[m][nt][r] + bv;
                v = fmaxf(v, 0.0f);
                tb[(size_t)p*CIN + ch] = (_Float16)v;
            }
        }
    }
}

// ---------------- Stage 4: 1x1 conv 64->256 + per-branch epilogue, WMMA -------------
// Wave: 2 M-tiles (32 pixels) x branch; A fragments loaded once, reused for 16 N-tiles.
__global__ __launch_bounds__(256) void k_conv1(const _Float16* __restrict__ t,
                                               const _Float16* __restrict__ wb1,
                                               const float* __restrict__ ob2,
                                               const float* __restrict__ sb2,
                                               float* __restrict__ offb,
                                               _Float16* __restrict__ scopeb) {
    int lane = threadIdx.x & 31, wave = threadIdx.x >> 5;
    int pairIdx = blockIdx.x * 4 + (wave >> 1);
    int branch  = wave & 1;
    int pix0 = pairIdx << 5;
    int row = lane & 15, kh = lane >> 4;

    const _Float16* src0 = t + (size_t)branch * T_HALVES + (size_t)(pix0 + row) * CIN;
    const _Float16* src1 = src0 + 16*CIN;
    AFrag a00, a01, a10, a11;           // [tile][chunk]
    a00.h[0] = *(const v8h*)(src0 +       kh*8);
    a00.h[1] = *(const v8h*)(src0 + 16 +  kh*8);
    a01.h[0] = *(const v8h*)(src0 + 32 +  kh*8);
    a01.h[1] = *(const v8h*)(src0 + 48 +  kh*8);
    a10.h[0] = *(const v8h*)(src1 +       kh*8);
    a10.h[1] = *(const v8h*)(src1 + 16 +  kh*8);
    a11.h[0] = *(const v8h*)(src1 + 32 +  kh*8);
    a11.h[1] = *(const v8h*)(src1 + 48 +  kh*8);

    const _Float16* wbase = wb1 + (size_t)branch * 16384;
    #pragma unroll 4
    for (int nt = 0; nt < 16; ++nt) {
        v16h b0 = *(const v16h*)(wbase + ((size_t)(nt*2 + 0)*32 + lane)*16);
        v16h b1 = *(const v16h*)(wbase + ((size_t)(nt*2 + 1)*32 + lane)*16);
        v8f acc0 = {}, acc1 = {};
        acc0 = __builtin_amdgcn_wmma_f32_16x16x32_f16(false, a00.v, false, b0, (short)0, acc0, false, false);
        acc0 = __builtin_amdgcn_wmma_f32_16x16x32_f16(false, a01.v, false, b1, (short)0, acc0, false, false);
        acc1 = __builtin_amdgcn_wmma_f32_16x16x32_f16(false, a10.v, false, b0, (short)0, acc1, false, false);
        acc1 = __builtin_amdgcn_wmma_f32_16x16x32_f16(false, a11.v, false, b1, (short)0, acc1, false, false);

        int ch = nt*16 + (lane & 15);
        if (branch == 0) {
            // offset = (conv + bias)*0.25 + INITIAL_POS
            float bv = ob2[ch];
            int d = ch >> 7, rem = ch & 127;
            float ip = (d == 0) ? ((rem & 1) ? 0.25f : -0.25f)
                                : (((rem >> 1) & 1) ? 0.25f : -0.25f);
            #pragma unroll
            for (int r = 0; r < 8; ++r) {
                int p0 = pix0 + r + 8*kh;
                offb[(size_t)p0*COUT + ch]        = (acc0[r] + bv)*0.25f + ip;
                offb[(size_t)(p0 + 16)*COUT + ch] = (acc1[r] + bv)*0.25f + ip;
            }
        } else {
            // scope = sigmoid(conv + bias)
            float bv = sb2[ch];
            #pragma unroll
            for (int r = 0; r < 8; ++r) {
                int p0 = pix0 + r + 8*kh;
                float s0 = 1.0f / (1.0f + __expf(-(acc0[r] + bv)));
                float s1 = 1.0f / (1.0f + __expf(-(acc1[r] + bv)));
                scopeb[(size_t)p0*COUT + ch]        = (_Float16)s0;
                scopeb[(size_t)(p0 + 16)*COUT + ch] = (_Float16)s1;
            }
        }
    }
}

// ---------------- Stage 5: pixel-shuffled grid-sample (border) * scope --------------
__global__ void k_sample(const float* __restrict__ x,
                         const float* __restrict__ offb,
                         const _Float16* __restrict__ scopeb,
                         float* __restrict__ out) {
    int idx = blockIdx.x * blockDim.x + threadIdx.x;     // (b, ch, oy, ox), ox fastest
    if (idx >= BB*64*512*512) return;
    int ox = idx & 511, oy = (idx >> 9) & 511, ch = (idx >> 18) & 63, b = idx >> 24;
    int h = oy >> 1, w = ox >> 1, r1 = oy & 1, r2 = ox & 1, gi = ch >> 1;

    int c0 = gi*8 + r1*2 + r2;   // grid x channel (pre-shuffle nc channel)
    int c1 = c0 + 4;             // grid y channel
    size_t pbase = ((size_t)(b << 16) + h*256 + w) * COUT;
    float offx = offb[pbase + c0];
    float offy = offb[pbase + c1];
    int d0 = c0 >> 7, d1 = c1 >> 7;
    float co0 = d0 ? (h + 0.5f) : (w + 0.5f);
    float co1 = d1 ? (h + 0.5f) : (w + 0.5f);
    float gx = 2.0f*(co0 + offx)*(1.0f/256.0f) - 1.0f;
    float gy = 2.0f*(co1 + offy)*(1.0f/256.0f) - 1.0f;

    float xs = fminf(fmaxf(((gx + 1.0f)*256.0f - 1.0f)*0.5f, 0.0f), 255.0f);
    float ys = fminf(fmaxf(((gy + 1.0f)*256.0f - 1.0f)*0.5f, 0.0f), 255.0f);
    float x0f = floorf(xs), y0f = floorf(ys);
    float wx = xs - x0f,   wy = ys - y0f;
    int x0 = (int)x0f, y0 = (int)y0f;
    int x1 = (x0 + 1 < 255) ? x0 + 1 : 255;
    int y1 = (y0 + 1 < 255) ? y0 + 1 : 255;

    const float* img = x + ((size_t)(b*64 + ch) << 16);
    float v00 = img[y0*256 + x0], v01 = img[y0*256 + x1];
    float v10 = img[y1*256 + x0], v11 = img[y1*256 + x1];
    float v = v00*(1.0f - wx)*(1.0f - wy) + v01*wx*(1.0f - wy)
            + v10*(1.0f - wx)*wy          + v11*wx*wy;

    // scope_r: flat reshape of NCHW (B,256,256,256) -> (B,64,512,512)
    int cs = ch*4 + (oy >> 7);
    int hp = ((oy & 127) << 1) + (ox >> 8);
    int wp = ox & 255;
    float sv = (float)scopeb[((size_t)(b << 16) + hp*256 + wp)*COUT + cs];

    out[idx] = v * sv;
}

extern "C" void kernel_launch(void* const* d_in, const int* in_sizes, int n_in,
                              void* d_out, int out_size, void* d_ws, size_t ws_size,
                              hipStream_t stream) {
    const float* x   = (const float*)d_in[0];
    const float* ow1 = (const float*)d_in[1];
    const float* ob1 = (const float*)d_in[2];
    const float* ow2 = (const float*)d_in[3];
    const float* ob2 = (const float*)d_in[4];
    const float* sw1 = (const float*)d_in[5];
    const float* sb1 = (const float*)d_in[6];
    const float* sw2 = (const float*)d_in[7];
    const float* sb2 = (const float*)d_in[8];
    float* out = (float*)d_out;

    char* ws = (char*)d_ws;
    _Float16* xpad   = (_Float16*)(ws + XPAD_OFF);
    _Float16* t      = (_Float16*)(ws + T_OFF);
    _Float16* wb3    = (_Float16*)(ws + WB3_OFF);
    _Float16* wb1    = (_Float16*)(ws + WB1_OFF);
    float*    offb   = (float*)   (ws + OFFB_OFF);
    _Float16* scopeb = (_Float16*)(ws + SCOPE_OFF);

    // Stage 0: zero padded buffer (halo)
    int n16 = (int)(XPAD_HALVES / 8);
    k_zero_xpad<<<(n16 + 255)/256, 256, 0, stream>>>((uint4*)xpad, n16);
    // Stage 1: convert/transpose x
    k_pack_x<<<(BB*CIN*HH*WW + 255)/256, 256, 0, stream>>>(x, xpad);
    // Stage 2: pack weights into WMMA B-fragment layout
    k_pack_w<<<(106496 + 255)/256, 256, 0, stream>>>(ow1, sw1, ow2, sw2, wb3, wb1);
    // Stage 3: 3x3 conv + ReLU (both branches), implicit GEMM via WMMA
    k_conv3<<<NPIX/32/4, 256, 0, stream>>>(xpad, wb3, ob1, sb1, t);
    // Stage 4: 1x1 conv + offset/scope epilogues via WMMA
    k_conv1<<<NPIX/32/4, 256, 0, stream>>>(t, wb1, ob2, sb2, offb, scopeb);
    // Stage 5: grid-sample * scope
    k_sample<<<(BB*64*512*512 + 255)/256, 256, 0, stream>>>(x, offb, scopeb, out);
}
